// HiddenWFACRF_27393301414309
// MI455X (gfx1250) — matrix operand order repaired and checked by reading, not compiled, not run on hardware
//
#include <hip/hip_runtime.h>
#include <hip/hip_bf16.h>

// ---------------------------------------------------------------------------
// HiddenWFACRF on gfx1250:
//   forward  : 16-WG cooperative scan, Em=exp(trans) f16 resident in LDS
//              (K-innermost layout, loaded by the Tensor Data Mover),
//              v_wmma_f32_16x16x32_f16 GEMM [32x64]@[64x8192] per step,
//              deterministic cross-WG ping-pong reduction published with
//              global_store_async_from_lds_b128 + spin barrier.
//   backward : (a) chunked parallel recompute of label-state logits with the
//              same WMMA GEMM (no sequential dependence once last[] stored),
//              (b) cheap sequential greedy decode, one WG per batch row.
// Workspace: Em(1MB) + last(8MB) + Hpart(256KB) + arrive(4KB) + ftb(8KB)
//            + chunk(64MB)  ~= 74 MB.
// ---------------------------------------------------------------------------

typedef __attribute__((ext_vector_type(16))) _Float16 v16h;
typedef __attribute__((ext_vector_type(8)))  float    v8f;
typedef __attribute__((ext_vector_type(4)))  unsigned v4u;
typedef __attribute__((ext_vector_type(8)))  int      v8i;
typedef __attribute__((ext_vector_type(4)))  int      v4i;

#define NB    32          // batch
#define NT    1024        // time
#define NL    128         // labels
#define NS    64          // states
#define NLS   8192        // L*S
#define NFWG  16          // forward workgroups (cooperating)
#define FCOLS 512         // columns of Em per forward WG
#define TC    64          // backward chunk length (time steps)
#define B1SL  8           // backward-logits column slices
#define B1C   1024        // columns per backward-logits WG
#define KSTR  72          // LDS pitch in halves for [j][k] Em tiles (144B = 9*16B)

#if __has_builtin(__builtin_amdgcn_tensor_load_to_lds)
#define HAVE_TDM 1
#else
#define HAVE_TDM 0
#endif

union UB { v16h v; v4u q[2]; _Float16 h[16]; };

// Fragment layouts (ISA 7.12.2, 16-bit, wave32):
//  A (16x32): lane m=lane&15; halves are k = kt*32 + (lane>>4)*8 + e  (e<8)
//             and k+16 for e>=8  -> two contiguous 8-half (16B) chunks.
//  B (32x16): lane n=lane&15; same k mapping as A -> two contiguous
//             8-half (16B) chunks.  With K-innermost LDS layout
//             both load as ds_load_b128 pairs.
__device__ __forceinline__ v16h load_a_frag(const _Float16* eaS, int lane, int mt, int kt) {
  UB u;
  const _Float16* src = eaS + (mt * 16 + (lane & 15)) * NS + kt * 32 + ((lane >> 4) << 3);
  u.q[0] = *(const v4u*)src;
  u.q[1] = *(const v4u*)(src + 16);
  return u.v;
}
__device__ __forceinline__ v16h load_b_frag(const _Float16* EmT, int lane, int jloc, int kt) {
  UB u;
  const _Float16* src = EmT + jloc * KSTR + kt * 32 + ((lane >> 4) << 3);
  u.q[0] = *(const v4u*)src;
  u.q[1] = *(const v4u*)(src + 16);
  return u.v;
}

// ---------------------------------------------------------------------------
// Tensor Data Mover: load `rows` rows of `tile0` f16 elements (contiguous in
// global) into LDS, inserting 4 DWORDs of padding every 32 DWORDs so that a
// 64-half logical row lands on a KSTR=72-half pitch.  D# per ISA ch.8.
// This toolchain's builtin takes 6 args (g0..g3 + extra int32x8 + cpol).
// ---------------------------------------------------------------------------
__device__ __forceinline__ void tdm_load_f16(unsigned lds_addr, const void* gsrc,
                                             unsigned tile0, unsigned rows) {
#if HAVE_TDM
  unsigned long long ga = (unsigned long long)(uintptr_t)gsrc;
  v4u g0; v8i g1; v4i g2 = {0, 0, 0, 0}, g3 = {0, 0, 0, 0};
  v8i g4 = {0, 0, 0, 0, 0, 0, 0, 0};
  g0[0] = 1u;                                      // count=1, user mode
  g0[1] = lds_addr;                                // lds_addr (bytes)
  g0[2] = (unsigned)(ga & 0xFFFFFFFFu);            // global_addr lo
  g0[3] = (unsigned)((ga >> 32) & 0x1FFFFFFu) | (2u << 30);  // ga hi | type=2
  g1[0] = (int)((1u << 16) |                       // data_size = 2 bytes
                (1u << 20) |                       // pad_enable
                (4u << 22) |                       // pad_interval: 32 DWORDs
                (3u << 25));                       // pad_amount : 4 DWORDs
  g1[1] = (int)((tile0 & 0xFFFFu) << 16);                        // tensor_dim0 lo16
  g1[2] = (int)(((tile0 >> 16) & 0xFFFFu) | ((rows & 0xFFFFu) << 16)); // hi16 | dim1 lo
  g1[3] = (int)(((rows >> 16) & 0xFFFFu) | ((tile0 & 0xFFFFu) << 16)); // dim1 hi | tile_dim0
  g1[4] = (int)(rows > 1 ? (rows & 0xFFFFu) : 0u); // tile_dim1 (0 => 1D)
  g1[5] = (int)tile0;                              // tensor_dim0_stride lo32
  g1[6] = 0; g1[7] = 0;
  __builtin_amdgcn_tensor_load_to_lds(g0, g1, g2, g3, g4, 0);
#else
  (void)lds_addr; (void)gsrc; (void)tile0; (void)rows;
#endif
}

// ---------------------------------------------------------------------------
// Kernel 0: EmT[j][k] = exp(transitions[k, j])  (K-innermost f16, j = l*64+s);
// zero the arrival counters.
// ---------------------------------------------------------------------------
__global__ __launch_bounds__(256) void k_prep(const float* __restrict__ tr,
                                              _Float16* __restrict__ EmT,
                                              int* __restrict__ arrive) {
  int i = blockIdx.x * 256 + threadIdx.x;
  if (i < NS * NLS) {
    int k = i >> 13, j = i & (NLS - 1);
    EmT[j * NS + k] = (_Float16)__expf(tr[i]);
  }
  if (i < NT) arrive[i] = 0;
}

// ---------------------------------------------------------------------------
// Kernel 1: forward scan. 16 WGs x 256 threads (8 wave32s).
// WG wg owns columns [wg*512, wg*512+512) (8 label blocks); wave wv owns
// label block wg*8+wv. Per step: EA[32x64] @ EmSlice -> w-weighted partial
// h[32x64], reduced across WGs via async-store slots + spin barrier.
// ---------------------------------------------------------------------------
__global__ __launch_bounds__(256) void k_fwd(const float* __restrict__ xl,
                                             const float* __restrict__ xs,
                                             const _Float16* __restrict__ EmG,
                                             float* __restrict__ lastG,
                                             float* __restrict__ Hpart,
                                             int* __restrict__ arrive) {
  extern __shared__ char smem[];
  _Float16* EmT  = (_Float16*)smem;                        // FCOLS*KSTR halves
  float*    lastS = (float*)(smem + FCOLS * KSTR * 2);     // 2048
  _Float16* eaS  = (_Float16*)((char*)lastS + 2048 * 4);   // 2048 halves
  float*    xlS  = (float*)((char*)eaS + 2048 * 2);        // 4096
  float*    hS   = xlS + 4096;                             // 2048
  float*    wS   = hS + 2048;                              // 256
  float*    m1S  = wS + 256;                               // 32
  float*    m2S  = m1S + 32;                               // 32
  float*    cS   = m2S + 32;                               // 32

  const int tid = threadIdx.x, lane = tid & 31, wv = tid >> 5;
  const int wg = blockIdx.x;
  const int col0 = wg * FCOLS;

  // Preload this WG's Em slice (K-innermost, contiguous in global) into LDS.
#if HAVE_TDM
  if (tid == 0)
    tdm_load_f16((unsigned)(uintptr_t)(void*)EmT, EmG + (size_t)col0 * NS,
                 FCOLS * NS, 1);
  __builtin_amdgcn_s_wait_tensorcnt(0);
#else
  for (int i = tid; i < FCOLS * NS; i += 256) {
    int j = i >> 6, k = i & 63;
    EmT[j * KSTR + k] = EmG[(size_t)col0 * NS + i];
  }
#endif

  // t = 0 : last0[b,s] = LSE_l(x_label[b,0,:]) + x_state[b,0,s]
  if (tid < NB) {
    const float* p = xl + (size_t)tid * NT * NL;
    float m = -1e30f;
    for (int l = 0; l < NL; l++) m = fmaxf(m, p[l]);
    float s = 0.f;
    for (int l = 0; l < NL; l++) s += __expf(p[l] - m);
    m1S[tid] = m + __logf(s);
  }
  __syncthreads();
  for (int i = tid; i < NB * NS; i += 256) {
    int b = i >> 6, s = i & 63;
    float v = m1S[b] + xs[((size_t)b * NT) * NS + s];
    lastS[i] = v;
    if (wg == 0) lastG[i] = v;
  }
  __syncthreads();

  for (int t = 1; t < NT; t++) {
    // ---- P0: load x_label[:,t,:], compute m1,m2, EA(f16), w, zero h ----
    for (int i = tid; i < NB * NL; i += 256)
      xlS[i] = xl[((size_t)(i >> 7) * NT + t) * NL + (i & 127)];
    for (int i = tid; i < NB * NS; i += 256) hS[i] = 0.f;
    __syncthreads();
    if (tid < NB) {
      int b = tid;
      float m1 = -1e30f, m2 = -1e30f;
      for (int s = 0; s < NS; s++) m1 = fmaxf(m1, lastS[b * NS + s]);
      for (int l = 0; l < NL; l++) m2 = fmaxf(m2, xlS[b * NL + l]);
      m1S[b] = m1; m2S[b] = m2; cS[b] = m1 + m2;
    }
    __syncthreads();
    for (int i = tid; i < NB * NS; i += 256)
      eaS[i] = (_Float16)__expf(lastS[i] - m1S[i >> 6]);
    {
      int b = tid >> 3, li = tid & 7;   // 256 threads == 32*8 exactly
      wS[tid] = __expf(xlS[b * NL + wg * 8 + li] - m2S[b]);
    }
    __syncthreads();

    // ---- P1: WMMA GEMM on this WG's slice; wave wv handles one l-block ----
    v16h a[2][2];
#pragma unroll
    for (int mt = 0; mt < 2; mt++)
#pragma unroll
      for (int kt = 0; kt < 2; kt++) a[mt][kt] = load_a_frag(eaS, lane, mt, kt);

    float wreg[2][8];                 // hoisted w[b, l=wg*8+wv] per acc row
#pragma unroll
    for (int mt = 0; mt < 2; mt++)
#pragma unroll
      for (int r = 0; r < 8; r++)
        wreg[mt][r] = wS[(mt * 16 + r + ((lane >> 4) << 3)) * 8 + wv];

    const int n = lane & 15;
#pragma unroll
    for (int st = 0; st < 4; st++) {
      int jloc = wv * 64 + st * 16 + n;
      v16h b0 = load_b_frag(EmT, lane, jloc, 0);
      v16h b1 = load_b_frag(EmT, lane, jloc, 1);
      int s = st * 16 + n;
#pragma unroll
      for (int mt = 0; mt < 2; mt++) {
        v8f acc = {0.f, 0.f, 0.f, 0.f, 0.f, 0.f, 0.f, 0.f};
        acc = __builtin_amdgcn_wmma_f32_16x16x32_f16(false, a[mt][0], false, b0,
                                                     (short)0, acc, false, false);
        acc = __builtin_amdgcn_wmma_f32_16x16x32_f16(false, a[mt][1], false, b1,
                                                     (short)0, acc, false, false);
#pragma unroll
        for (int r = 0; r < 8; r++) {
          int b = mt * 16 + r + ((lane >> 4) << 3);
          atomicAdd(&hS[b * NS + s], wreg[mt][r] * acc[r]);  // ds_add_f32
        }
      }
    }
    __syncthreads();

    // ---- P2: publish partial h (LDS -> global, async), barrier, combine ----
    float* slot = Hpart + ((size_t)(t & 1) * NFWG + wg) * (NB * NS);
    {
      unsigned lsrc = (unsigned)(uintptr_t)(void*)hS + (unsigned)tid * 32u;
      unsigned long long gd = (unsigned long long)(uintptr_t)(void*)slot +
                              (unsigned long long)tid * 32ull;
      asm volatile("global_store_async_from_lds_b128 %0, %1, off scope:SCOPE_DEV"
                   :: "v"(gd), "v"(lsrc) : "memory");
      asm volatile("global_store_async_from_lds_b128 %0, %1, off scope:SCOPE_DEV"
                   :: "v"(gd + 16ull), "v"(lsrc + 16u) : "memory");
      asm volatile("s_wait_asynccnt 0" ::: "memory");
    }
    __threadfence();
    __syncthreads();
    if (tid == 0) {
      __hip_atomic_fetch_add(&arrive[t], 1, __ATOMIC_RELEASE, __HIP_MEMORY_SCOPE_AGENT);
      while (__hip_atomic_load(&arrive[t], __ATOMIC_ACQUIRE, __HIP_MEMORY_SCOPE_AGENT) < NFWG)
        __builtin_amdgcn_s_sleep(1);
    }
    __syncthreads();
    const float* base = Hpart + (size_t)(t & 1) * NFWG * NB * NS;
    for (int i = tid; i < NB * NS; i += 256) {
      float h = 0.f;
      for (int sl = 0; sl < NFWG; sl++)   // fixed order => deterministic
        h += __hip_atomic_load(&base[sl * NB * NS + i], __ATOMIC_RELAXED,
                               __HIP_MEMORY_SCOPE_AGENT);
      int b = i >> 6, s = i & 63;
      float v = xs[((size_t)b * NT + t) * NS + s] + cS[b] + __logf(h);
      lastS[i] = v;
      if (wg == 0) lastG[(size_t)t * NB * NS + i] = v;
    }
    __syncthreads();
  }
}

// ---------------------------------------------------------------------------
// Kernel 2: backward logits for one 64-step chunk (fully parallel WMMA).
// chunk[t'][b][j] = lsi_t[b, l, s] = xl + xs + m1 + log f   (f via GEMM)
// ---------------------------------------------------------------------------
__global__ __launch_bounds__(256) void k_bwd_logits(const float* __restrict__ xl,
                                                    const float* __restrict__ xs,
                                                    const _Float16* __restrict__ EmG,
                                                    const float* __restrict__ lastG,
                                                    float* __restrict__ chunk,
                                                    int t0) {
  extern __shared__ char smem[];
  _Float16* EmT = (_Float16*)smem;                     // B1C*KSTR halves
  float*    xlS = (float*)(smem + B1C * KSTR * 2);     // 32*16
  float*    xsS = xlS + 32 * 16;                       // 2048
  _Float16* eaS = (_Float16*)(xsS + 2048);             // 2048 halves
  float*    m1S = (float*)((char*)eaS + 2048 * 2);     // 32

  const int tid = threadIdx.x, lane = tid & 31, wv = tid >> 5;
  const int t1 = blockIdx.x / B1SL, slice = blockIdx.x % B1SL;
  const int t = t0 + t1;
  const int col0 = slice * B1C;

  if (t == 0) {  // lsl[0] = base (no transition term)
    for (int i = tid; i < NB * B1C; i += 256) {
      int b = i >> 10, j = col0 + (i & 1023);
      int l = j >> 6, s = j & 63;
      chunk[((size_t)t1 * NB + b) * NLS + j] =
          xl[((size_t)b * NT + 0) * NL + l] + xs[((size_t)b * NT + 0) * NS + s];
    }
    return;
  }

#if HAVE_TDM
  if (tid == 0)
    tdm_load_f16((unsigned)(uintptr_t)(void*)EmT, EmG + (size_t)col0 * NS,
                 (B1C * NS) / 2, 2);
  __builtin_amdgcn_s_wait_tensorcnt(0);
#else
  for (int i = tid; i < B1C * NS; i += 256) {
    int j = i >> 6, k = i & 63;
    EmT[j * KSTR + k] = EmG[(size_t)col0 * NS + i];
  }
#endif
  for (int i = tid; i < NB * 16; i += 256)
    xlS[i] = xl[((size_t)(i >> 4) * NT + t) * NL + slice * 16 + (i & 15)];
  for (int i = tid; i < NB * NS; i += 256)
    xsS[i] = xs[((size_t)(i >> 6) * NT + t) * NS + (i & 63)];
  __syncthreads();
  if (tid < NB) {
    const float* lp = lastG + (size_t)(t - 1) * NB * NS + tid * NS;
    float m = -1e30f;
    for (int s = 0; s < NS; s++) m = fmaxf(m, lp[s]);
    m1S[tid] = m;
  }
  __syncthreads();
  for (int i = tid; i < NB * NS; i += 256)
    eaS[i] = (_Float16)__expf(lastG[(size_t)(t - 1) * NB * NS + i] - m1S[i >> 6]);
  __syncthreads();

  v16h a[2][2];
#pragma unroll
  for (int mt = 0; mt < 2; mt++)
#pragma unroll
    for (int kt = 0; kt < 2; kt++) a[mt][kt] = load_a_frag(eaS, lane, mt, kt);

  const int n = lane & 15;
  for (int p = 0; p < 2; p++) {        // two l-blocks per wave
    int lb = wv * 2 + p;               // local l index (0..15)
#pragma unroll
    for (int st = 0; st < 4; st++) {
      int jloc = lb * 64 + st * 16 + n;
      v16h b0 = load_b_frag(EmT, lane, jloc, 0);
      v16h b1 = load_b_frag(EmT, lane, jloc, 1);
      int s = st * 16 + n;
      int j = col0 + jloc;
#pragma unroll
      for (int mt = 0; mt < 2; mt++) {
        v8f acc = {0.f, 0.f, 0.f, 0.f, 0.f, 0.f, 0.f, 0.f};
        acc = __builtin_amdgcn_wmma_f32_16x16x32_f16(false, a[mt][0], false, b0,
                                                     (short)0, acc, false, false);
        acc = __builtin_amdgcn_wmma_f32_16x16x32_f16(false, a[mt][1], false, b1,
                                                     (short)0, acc, false, false);
#pragma unroll
        for (int r = 0; r < 8; r++) {
          int b = mt * 16 + r + ((lane >> 4) << 3);
          chunk[((size_t)t1 * NB + b) * NLS + j] =
              xlS[b * 16 + lb] + xsS[b * NS + s] + m1S[b] + __logf(acc[r]);
        }
      }
    }
  }
}

// ---------------------------------------------------------------------------
// Kernel 3: sequential greedy decode within a chunk, one WG per batch row.
// Uses the exact -10000-mask collapse and the reference's trans_b index
// pairing: ft[s0] = LSE_s( sll[l*,s] + trans[s0, 64*(l*&1)+s, l*>>1] ).
// ---------------------------------------------------------------------------
__global__ __launch_bounds__(256) void k_bwd_decode(const float* __restrict__ tr,
                                                    const float* __restrict__ chunk,
                                                    float* __restrict__ ftbG,
                                                    float* __restrict__ out,
                                                    int t0, int isLast) {
  __shared__ float ftb[NS];
  __shared__ float R[NL];
  __shared__ float redV[NL];
  __shared__ int   redI[NL];
  __shared__ float rowS[NS];
  __shared__ int   lstarS;

  const int b = blockIdx.x;
  const int tid = threadIdx.x, lane = tid & 31, wv = tid >> 5;

  if (tid < NS) ftb[tid] = isLast ? 0.f : ftbG[b * NS + tid];
  __syncthreads();

  for (int t1 = TC - 1; t1 >= 0; t1--) {
    const int t = t0 + t1;
    const float* P = chunk + ((size_t)t1 * NB + b) * NLS;

    // R[l] = LSE_s( P[l,s] + ftb[s] )  — 16 labels per wave, wave32 shuffles
    for (int li = 0; li < 16; li++) {
      int l = wv * 16 + li;
      float v0 = P[l * NS + lane] + ftb[lane];
      float v1 = P[l * NS + 32 + lane] + ftb[32 + lane];
      float m = fmaxf(v0, v1);
      for (int o = 16; o >= 1; o >>= 1) m = fmaxf(m, __shfl_xor(m, o, 32));
      float e = __expf(v0 - m) + __expf(v1 - m);
      for (int o = 16; o >= 1; o >>= 1) e += __shfl_xor(e, o, 32);
      if (lane == 0) R[l] = m + __logf(e);
    }
    __syncthreads();

    // argmax over labels, first-index tie-break (matches jnp.argmax)
    if (tid < NL) { redV[tid] = R[tid]; redI[tid] = tid; }
    __syncthreads();
    for (int off = NL / 2; off >= 1; off >>= 1) {
      if (tid < off) {
        float av = redV[tid], cv = redV[tid + off];
        int   ai = redI[tid], ci = redI[tid + off];
        if (cv > av || (cv == av && ci < ai)) { redV[tid] = cv; redI[tid] = ci; }
      }
      __syncthreads();
    }
    if (tid == 0) { lstarS = redI[0]; out[(size_t)b * NT + t] = (float)redI[0]; }
    __syncthreads();
    const int ls = lstarS;
    if (tid < NS) rowS[tid] = P[ls * NS + tid] + ftb[tid];
    __syncthreads();

    // ftb update for the next (earlier) step
    if (tid < NS) {
      int s0 = tid;
      int lp0 = (ls & 1) * 64;
      int sp = ls >> 1;
      const float* tp = tr + (size_t)s0 * NLS + sp;
      float m = -1e30f;
      for (int s = 0; s < NS; s++) m = fmaxf(m, rowS[s] + tp[(lp0 + s) * NS]);
      float e = 0.f;
      for (int s = 0; s < NS; s++) e += __expf(rowS[s] + tp[(lp0 + s) * NS] - m);
      redV[s0] = m + __logf(e);
    }
    __syncthreads();
    if (tid < NS) ftb[tid] = redV[tid];
    __syncthreads();
  }
  if (tid < NS) ftbG[b * NS + tid] = ftb[tid];
}

// ---------------------------------------------------------------------------
extern "C" void kernel_launch(void* const* d_in, const int* in_sizes, int n_in,
                              void* d_out, int out_size, void* d_ws, size_t ws_size,
                              hipStream_t stream) {
  const float* xl = (const float*)d_in[0];   // [B,T,L]
  const float* xs = (const float*)d_in[1];   // [B,T,S]
  const float* tr = (const float*)d_in[2];   // [S,L,S]
  float* out = (float*)d_out;                // [B,T]

  char* w = (char*)d_ws;
  size_t off = 0;
  _Float16* Em    = (_Float16*)(w + off); off += (size_t)NS * NLS * 2;        // 1 MB
  float*    lastG = (float*)(w + off);    off += (size_t)NT * NB * NS * 4;    // 8 MB
  float*    Hpart = (float*)(w + off);    off += (size_t)2 * NFWG * NB * NS * 4;
  int*      arrive= (int*)(w + off);      off += (size_t)NT * 4;
  float*    ftbG  = (float*)(w + off);    off += (size_t)NB * NS * 4;
  float*    chunk = (float*)(w + off);    off += (size_t)TC * NB * NLS * 4;   // 64 MB

  const size_t fwd_lds = (size_t)FCOLS * KSTR * 2 + 2048 * 4 + 2048 * 2 +
                         4096 * 4 + 2048 * 4 + 256 * 4 + 3 * 32 * 4;
  const size_t b1_lds  = (size_t)B1C * KSTR * 2 + 32 * 16 * 4 + 2048 * 4 +
                         2048 * 2 + 32 * 4;

  k_prep<<<(NS * NLS) / 256, 256, 0, stream>>>(tr, Em, arrive);
  k_fwd<<<NFWG, 256, fwd_lds, stream>>>(xl, xs, Em, lastG, Hpart, arrive);

  const int NCHUNK = NT / TC;
  for (int c = NCHUNK - 1; c >= 0; c--) {
    k_bwd_logits<<<TC * B1SL, 256, b1_lds, stream>>>(xl, xs, Em, lastG, chunk, c * TC);
    k_bwd_decode<<<NB, 256, 0, stream>>>(tr, chunk, ftbG, out, c * TC, c == NCHUNK - 1);
  }
  (void)in_sizes; (void)n_in; (void)out_size; (void)ws_size;
}
